// RNNCategoricalActor_DSE_76235669504094
// MI455X (gfx1250) — matrix operand order repaired and checked by reading, not compiled, not run on hardware
//
#include <hip/hip_runtime.h>

// ---------------- problem constants ----------------
#define BTOT 32768
#define DIN  16
#define HD   128
#define FD   64
#define NH   16
#define NA   32

#define BT   128            // samples per workgroup
#define KX   160            // padded input K (16 obs + 1 idx + 15 pad + 128 h0)
#define SXK  168            // LDS row stride (bf16) for X and Wcat: 16B aligned, bank-spread
#define SHK  136            // LDS row stride (bf16) for h1 and W1
#define SO2  68             // LDS row stride (f32) for out2
#define SLG  33             // LDS row stride (f32) for logits

typedef __attribute__((ext_vector_type(16))) __bf16        v16bf;
typedef __attribute__((ext_vector_type(8)))  float         v8f;
typedef __attribute__((ext_vector_type(4)))  unsigned int  u32x4;
typedef __attribute__((ext_vector_type(4)))  float         f32x4;

union FragU { u32x4 q[2]; v16bf v; };

static __device__ __forceinline__ v16bf load_frag(const unsigned short* p0,
                                                  const unsigned short* p1) {
    FragU u;
    u.q[0] = *(const u32x4*)p0;   // 16B LDS read
    u.q[1] = *(const u32x4*)p1;   // 16B LDS read
    return u.v;
}

static __device__ __forceinline__ unsigned short f2bf(float f) {
    unsigned int u = __float_as_uint(f);
    u += 0x7FFFu + ((u >> 16) & 1u);          // round-to-nearest-even
    return (unsigned short)(u >> 16);
}
static __device__ __forceinline__ float bf2f(unsigned short b) {
    return __uint_as_float(((unsigned int)b) << 16);
}
static __device__ __forceinline__ float sigmoidf_(float x) {
    return 1.0f / (1.0f + __expf(-x));
}

// -------- LDS layout: Wcat region is dead after gates -> overlay head data --------
struct SMemP2 {
    float          out2[BT * SO2];        // 34816 B
    float          logits[BT * SLG];      // 16896 B
    unsigned short w2[NH * FD * NA];      // 65536 B   [head][k][a]
    float          b2[NH * NA];           //  2048 B
};
struct SMem {
    union {
        unsigned short wg[512 * SXK];     // 172032 B  [n][k] bf16 (W_ih|0|W_hh)
        SMemP2 p2;                        // 119296 B
    } ov;
    unsigned short x[BT * SXK];           //  43008 B  padded inputs, bf16
    unsigned short w1[FD * SHK];          //  17408 B  [n][k] bf16
    unsigned short h1[BT * SHK];          //  34816 B  bf16 h1 for fc1
    float          bg[512];               //   2048 B  b_ih + b_hh
    float          b1f[FD];               //    256 B
};                                        // total 269568 B  (<= 320KB WGP LDS)

__global__ __launch_bounds__(256, 1)
void rnn_actor_wmma_kernel(const float* __restrict__ obs,
                           const int*   __restrict__ act_idx,
                           const float* __restrict__ h0,
                           const float* __restrict__ c0,
                           const int*   __restrict__ act,
                           const float* __restrict__ W_ih,
                           const float* __restrict__ W_hh,
                           const float* __restrict__ b_ih,
                           const float* __restrict__ b_hh,
                           const float* __restrict__ W1,
                           const float* __restrict__ b1,
                           const float* __restrict__ W2,
                           const float* __restrict__ b2,
                           float* __restrict__ o_logits,
                           float* __restrict__ o_logpa,
                           float* __restrict__ o_c1_out,
                           float* __restrict__ o_h1_out) {
    __shared__ SMem sm;

    const int tid  = threadIdx.x;
    const int s0   = blockIdx.x * BT;
    const int w    = tid >> 5;        // wave id 0..7
    const int lane = tid & 31;
    const int lrow = lane & 15;
    const int half = lane >> 4;

    // -------------------- stage 1: inputs + gate weights + fc1 weights --------------------
    // warm W2 (head phase) and this block's c0 tile (LSTM phase) into cache
    if (tid < 128) __builtin_prefetch(W2 + tid * 256, 0, 1);          // 128KB of W2
    __builtin_prefetch(c0 + (size_t)s0 * HD + tid * 64, 0, 1);        // 64KB c0 tile

    // X = [obs(16) | idx/16 | zeros(15) | h0(128) | zeros(pad)]  as bf16
    for (int e = tid; e < BT * SXK; e += 256) {
        int s = e / SXK, c = e - s * SXK;
        float v = 0.0f;
        if (c < DIN)                 v = obs[(s0 + s) * DIN + c];
        else if (c == DIN)           v = (float)act_idx[s0 + s] * (1.0f / (float)NH);
        else if (c >= 32 && c < KX)  v = h0[(size_t)(s0 + s) * HD + (c - 32)];
        sm.x[e] = f2bf(v);
    }
    // Wcat[n][k] : k<17 -> W_ih[n][k] ; 32<=k<160 -> W_hh[n][k-32] ; else 0
    for (int e = tid; e < 512 * SXK; e += 256) {
        int n = e / SXK, k = e - n * SXK;
        float v = 0.0f;
        if (k < DIN + 1)             v = W_ih[n * (DIN + 1) + k];
        else if (k >= 32 && k < KX)  v = W_hh[n * HD + (k - 32)];
        sm.ov.wg[e] = f2bf(v);
    }
    for (int e = tid; e < FD * SHK; e += 256) {
        int n = e / SHK, k = e - n * SHK;
        sm.w1[e] = f2bf(k < HD ? W1[n * HD + k] : 0.0f);
    }
    for (int e = tid; e < 512; e += 256) sm.bg[e] = b_ih[e] + b_hh[e];
    if (tid < FD) sm.b1f[tid] = b1[tid];
    __syncthreads();

    // -------------------- stage 2: gate GEMM (bf16 WMMA) + LSTM elementwise --------------------
    // wave w owns h-channels [16w, 16w+16); gate g lives at N-tile g*8 + w.
    // Two sample tiles per iteration (B fragment feeds 2 independent chains),
    // kc-outer / g-inner order -> accumulator reuse distance = 8 WMMAs, so the
    // bf16 WMMA->WMMA RAW hazard window (5 slots) is filled with real work.
    const int ch = 16 * w + lrow;
    const unsigned short* pBrow[4];
    #pragma unroll
    for (int g = 0; g < 4; ++g)
        pBrow[g] = sm.ov.wg + (128 * g + 16 * w + lrow) * SXK + 8 * half;

    for (int mt = 0; mt < 8; mt += 2) {
        const unsigned short* pA0 = sm.x + ((mt + 0) * 16 + lrow) * SXK + 8 * half;
        const unsigned short* pA1 = sm.x + ((mt + 1) * 16 + lrow) * SXK + 8 * half;
        v16bf afr0[5], afr1[5];
        #pragma unroll
        for (int kc = 0; kc < 5; ++kc) {
            afr0[kc] = load_frag(pA0 + 32 * kc, pA0 + 32 * kc + 16);
            afr1[kc] = load_frag(pA1 + 32 * kc, pA1 + 32 * kc + 16);
        }

        v8f acc0[4], acc1[4];
        #pragma unroll
        for (int g = 0; g < 4; ++g) {
            float bias = sm.bg[128 * g + 16 * w + lrow];
            #pragma unroll
            for (int e = 0; e < 8; ++e) { acc0[g][e] = bias; acc1[g][e] = bias; }
        }
        #pragma unroll
        for (int kc = 0; kc < 5; ++kc) {
            v16bf bfr[4];
            #pragma unroll
            for (int g = 0; g < 4; ++g)
                bfr[g] = load_frag(pBrow[g] + 32 * kc, pBrow[g] + 32 * kc + 16);
            #pragma unroll
            for (int g = 0; g < 4; ++g) {
                acc0[g] = __builtin_amdgcn_wmma_f32_16x16x32_bf16(
                              false, afr0[kc], false, bfr[g], (short)0, acc0[g], false, false);
                acc1[g] = __builtin_amdgcn_wmma_f32_16x16x32_bf16(
                              false, afr1[kc], false, bfr[g], (short)0, acc1[g], false, false);
            }
        }
        // LSTM cell on the C-fragment layout: VGPR r <-> row base + r + 8*half
        #pragma unroll
        for (int t = 0; t < 2; ++t) {
            #pragma unroll
            for (int r = 0; r < 8; ++r) {
                int srow = (mt + t) * 16 + r + 8 * half;
                int s    = s0 + srow;
                float iv = t ? acc1[0][r] : acc0[0][r];
                float fv = t ? acc1[1][r] : acc0[1][r];
                float gv = t ? acc1[2][r] : acc0[2][r];
                float ov = t ? acc1[3][r] : acc0[3][r];
                float c0v = c0[(size_t)s * HD + ch];
                float c1v = sigmoidf_(fv) * c0v + sigmoidf_(iv) * tanhf(gv);
                float h1v = sigmoidf_(ov) * tanhf(c1v);
                o_c1_out[(size_t)s * HD + ch] = c1v;
                o_h1_out[(size_t)s * HD + ch] = h1v;
                sm.h1[srow * SHK + ch] = f2bf(h1v);
            }
        }
    }
    __syncthreads();   // Wcat region now dead -> overlay becomes live

    // -------------------- stage 3: stage routed-head weights + fc1 WMMA --------------------
    for (int e = tid; e < NH * NA * FD; e += 256) {
        int h = e >> 11, rem = e & 2047;
        int a = rem >> 6, k = rem & 63;
        sm.ov.p2.w2[(h * FD + k) * NA + a] = f2bf(W2[e]);   // [head][k][a]
    }
    for (int e = tid; e < NH * NA; e += 256) sm.ov.p2.b2[e] = b2[e];

    {   // out2 = relu(h1 @ W1^T + b1) ; wave w owns samples [16w, 16w+16)
        // kc-outer / nt-inner: 4 live accumulators, reuse distance 4.
        const unsigned short* pA = sm.h1 + (w * 16 + lrow) * SHK + 8 * half;
        v16bf afr[4];
        #pragma unroll
        for (int kc = 0; kc < 4; ++kc)
            afr[kc] = load_frag(pA + 32 * kc, pA + 32 * kc + 16);

        v8f acc[4];
        #pragma unroll
        for (int nt = 0; nt < 4; ++nt) {
            float bias = sm.b1f[nt * 16 + lrow];
            #pragma unroll
            for (int e = 0; e < 8; ++e) acc[nt][e] = bias;
        }
        const unsigned short* pB1row[4];
        #pragma unroll
        for (int nt = 0; nt < 4; ++nt)
            pB1row[nt] = sm.w1 + (nt * 16 + lrow) * SHK + 8 * half;

        #pragma unroll
        for (int kc = 0; kc < 4; ++kc) {
            v16bf bfr[4];
            #pragma unroll
            for (int nt = 0; nt < 4; ++nt)
                bfr[nt] = load_frag(pB1row[nt] + 32 * kc, pB1row[nt] + 32 * kc + 16);
            #pragma unroll
            for (int nt = 0; nt < 4; ++nt)
                acc[nt] = __builtin_amdgcn_wmma_f32_16x16x32_bf16(
                              false, afr[kc], false, bfr[nt], (short)0, acc[nt], false, false);
        }
        #pragma unroll
        for (int nt = 0; nt < 4; ++nt) {
            #pragma unroll
            for (int r = 0; r < 8; ++r) {
                float v = acc[nt][r];
                v = v > 0.0f ? v : 0.0f;
                sm.ov.p2.out2[(w * 16 + r + 8 * half) * SO2 + nt * 16 + lrow] = v;
            }
        }
    }
    __syncthreads();

    // -------------------- stage 4: routed head (per-sample gathered W2, VALU) --------------------
    // wave = one sample at a time, lane = action
    for (int j = 0; j < 16; ++j) {
        int sl = w * 16 + j;
        int s  = s0 + sl;
        int hh = act_idx[s];
        float accv = sm.ov.p2.b2[hh * NA + lane];
        const unsigned short* w2p = sm.ov.p2.w2 + hh * FD * NA + lane;
        const float* o2 = sm.ov.p2.out2 + sl * SO2;
        #pragma unroll 4
        for (int k = 0; k < FD; k += 4) {
            f32x4 ov = *(const f32x4*)(o2 + k);              // broadcast 16B LDS read
            accv += ov[0] * bf2f(w2p[(k + 0) * NA]);
            accv += ov[1] * bf2f(w2p[(k + 1) * NA]);
            accv += ov[2] * bf2f(w2p[(k + 2) * NA]);
            accv += ov[3] * bf2f(w2p[(k + 3) * NA]);
        }
        float lg = tanhf(accv);
        o_logits[(size_t)s * NA + lane] = lg;                 // coalesced
        sm.ov.p2.logits[sl * SLG + lane] = lg;
    }
    __syncthreads();

    // -------------------- stage 5: log_softmax gather --------------------
    if (tid < BT) {
        const float* L = sm.ov.p2.logits + tid * SLG;
        float m = L[0];
        #pragma unroll
        for (int i = 1; i < NA; ++i) m = fmaxf(m, L[i]);
        float ssum = 0.0f;
        #pragma unroll
        for (int i = 0; i < NA; ++i) ssum += __expf(L[i] - m);
        int s  = s0 + tid;
        int av = act[s];
        o_logpa[s] = (L[av] - m) - __logf(ssum);
    }
}

extern "C" void kernel_launch(void* const* d_in, const int* in_sizes, int n_in,
                              void* d_out, int out_size, void* d_ws, size_t ws_size,
                              hipStream_t stream) {
    (void)in_sizes; (void)n_in; (void)out_size; (void)d_ws; (void)ws_size;
    const float* obs     = (const float*)d_in[0];
    const int*   act_idx = (const int*)  d_in[1];
    const float* h0      = (const float*)d_in[2];
    const float* c0      = (const float*)d_in[3];
    const int*   act     = (const int*)  d_in[4];
    const float* W_ih    = (const float*)d_in[5];
    const float* W_hh    = (const float*)d_in[6];
    const float* b_ih    = (const float*)d_in[7];
    const float* b_hh    = (const float*)d_in[8];
    const float* W1      = (const float*)d_in[9];
    const float* b1      = (const float*)d_in[10];
    const float* W2      = (const float*)d_in[11];
    const float* b2      = (const float*)d_in[12];

    float* out      = (float*)d_out;
    float* o_logits = out;                                    // [B, 32]
    float* o_logpa  = out + (size_t)BTOT * 32;                // [B]
    float* o_h1     = out + (size_t)BTOT * 33;                // [B, 128]
    float* o_c1     = out + (size_t)BTOT * 33 + (size_t)BTOT * 128;  // [B, 128]

    dim3 grid(BTOT / BT), block(256);
    rnn_actor_wmma_kernel<<<grid, block, 0, stream>>>(
        obs, act_idx, h0, c0, act, W_ih, W_hh, b_ih, b_hh, W1, b1, W2, b2,
        o_logits, o_logpa, o_c1, o_h1);
}
